// TransformerLayer_7885559955613
// MI455X (gfx1250) — matrix operand order repaired and checked by reading, not compiled, not run on hardware
//
#include <hip/hip_runtime.h>
#include <math.h>
#include <stdint.h>

typedef __attribute__((ext_vector_type(16))) __bf16 v16bf;
typedef __attribute__((ext_vector_type(8)))  float  v8f;
typedef __attribute__((ext_vector_type(4)))  unsigned int u32x4;
typedef unsigned short u16;
typedef unsigned int   u32;

#define HD __device__ __forceinline__

static constexpr int Bsz = 8, Lseq = 1024, Dm = 768, Hh = 12, DHd = 64, HIDd = 3072;

HD u16 f2bf(float f) {
    u32 u = __float_as_uint(f);
    u32 r = u + 0x7FFFu + ((u >> 16) & 1u);
    return (u16)(r >> 16);
}

struct Frag { u32x4 u[2]; };          // 32 bytes: one 16xbf16 WMMA operand
HD v16bf asbf(const Frag& f) { return __builtin_bit_cast(v16bf, f); }
HD v8f vzero() { v8f z = {0.f,0.f,0.f,0.f,0.f,0.f,0.f,0.f}; return z; }

// Flat shared-pointer -> 32-bit LDS byte address. Per CDNA5 aperture rules the
// LDS aperture tag lives entirely in addr[63:32]; the low 32 bits are the LDS byte address.
HD u32 lds_addr(const void* p) { return (u32)(uintptr_t)p; }

// Issue a 32-byte async DMA (two b128) global->LDS. offset:16 advances BOTH
// the global and LDS addresses.
HD void async_copy32(u32 lds, const u16* gaddr) {
    asm volatile(
        "global_load_async_to_lds_b128 %0, %1, off\n\t"
        "global_load_async_to_lds_b128 %0, %1, off offset:16"
        :: "v"(lds), "v"(gaddr) : "memory");
}
HD void wait_async0() { asm volatile("s_wait_asynccnt 0" ::: "memory"); }

// ---------------------------------------------------------------------------
// Weight transpose + fp32 -> bf16 convert:  W[K][N] -> Wt[N][K]
// ---------------------------------------------------------------------------
__global__ void cvt_transpose_kernel(const float* __restrict__ W,
                                     u16* __restrict__ Wt, int K, int N) {
    int i = blockIdx.x * 256 + threadIdx.x;
    if (i >= K * N) return;
    int n = i / K, k = i % K;
    Wt[i] = f2bf(W[(size_t)k * N + n]);
}

// ---------------------------------------------------------------------------
// LayerNorm over D=768, fp32 in -> bf16 out. One 256-thread block per row.
// ---------------------------------------------------------------------------
__global__ __launch_bounds__(256) void ln_kernel(const float* __restrict__ X,
                                                 const float* __restrict__ g,
                                                 const float* __restrict__ b,
                                                 u16* __restrict__ Y) {
    __shared__ float r1[256], r2[256];
    const int row = blockIdx.x, tid = threadIdx.x;
    const float* xr = X + (size_t)row * Dm;
    float x0 = xr[tid], x1 = xr[tid + 256], x2 = xr[tid + 512];
    r1[tid] = x0 + x1 + x2;
    r2[tid] = x0 * x0 + x1 * x1 + x2 * x2;
    __syncthreads();
    for (int off = 128; off > 0; off >>= 1) {
        if (tid < off) { r1[tid] += r1[tid + off]; r2[tid] += r2[tid + off]; }
        __syncthreads();
    }
    const float mu = r1[0] * (1.f / Dm);
    const float var = r2[0] * (1.f / Dm) - mu * mu;
    const float rs = rsqrtf(var + 1e-6f);
    u16* yr = Y + (size_t)row * Dm;
    yr[tid]       = f2bf((x0 - mu) * rs * g[tid]       + b[tid]);
    yr[tid + 256] = f2bf((x1 - mu) * rs * g[tid + 256] + b[tid + 256]);
    yr[tid + 512] = f2bf((x2 - mu) * rs * g[tid + 512] + b[tid + 512]);
}

// ---------------------------------------------------------------------------
// Generic bf16 WMMA GEMM: C[M,N] = A[M,K] * B[K,N], B pre-transposed Bt[N][K].
// 128x128x32 tiles, 8 waves (4 row x 2 col), each wave 32x64 = 2x4 frags.
// Double-buffered async DMA staging: tile t+1 streams into LDS while tile t
// feeds the WMMAs; one barrier + one s_wait_asynccnt per k-step.
// ---------------------------------------------------------------------------
enum { EPI_QKV = 0, EPI_PROJ = 1, EPI_FC1 = 2, EPI_FC2 = 3 };
static constexpr int GT = 128 * 40;   // one 128x32 tile (+8 u16 row pad)

template <int EPI>
__global__ __launch_bounds__(256) void gemm_bf16_kernel(
    const u16* __restrict__ A, const u16* __restrict__ Bt,
    int M, int N, int K,
    const float* __restrict__ bias, const float* __restrict__ resid,
    float* __restrict__ fout, u16* __restrict__ hout,
    u16* __restrict__ qOut, u16* __restrict__ kOut, u16* __restrict__ vtOut) {
    __shared__ u16 As[2 * GT];
    __shared__ u16 Bs[2 * GT];
    const int tid = threadIdx.x;
    const int wave = tid >> 5, lane = tid & 31;
    const int lrow = lane & 15, hi = lane >> 4;
    const int rw = wave >> 1, cw = wave & 1;
    const int m0 = blockIdx.y * 128, n0 = blockIdx.x * 128;
    const int ldr = tid >> 1, ldh = tid & 1;

    const u32 dA = lds_addr(As + ldr * 40 + ldh * 16);
    const u32 dB = lds_addr(Bs + ldr * 40 + ldh * 16);
    auto issue = [&](int k0, int buf) {
        const u16* pa = A + (size_t)(m0 + ldr) * K + k0 + ldh * 16;
        const u16* pb = Bt + (size_t)(n0 + ldr) * K + k0 + ldh * 16;
        async_copy32(dA + buf * (GT * 2), pa);
        async_copy32(dB + buf * (GT * 2), pb);
        if (k0 + 32 < K) {               // warm tile t+2 into L2
            __builtin_prefetch(pa + 32, 0, 0);
            __builtin_prefetch(pb + 32, 0, 0);
        }
    };

    v8f C[2][4];
#pragma unroll
    for (int i = 0; i < 2; ++i)
#pragma unroll
        for (int j = 0; j < 4; ++j) C[i][j] = vzero();

    const int nt = K / 32;
    issue(0, 0);
    for (int t = 0; t < nt; ++t) {
        const int p = t & 1;
        wait_async0();        // our DMA for tile t has landed in LDS
        __syncthreads();      // everyone's has; prior tile fully consumed
        if (t + 1 < nt) issue((t + 1) * 32, 1 - p);

        const u16* as = As + p * GT;
        const u16* bs = Bs + p * GT;
        Frag Af[2], Bf[4];
#pragma unroll
        for (int mi = 0; mi < 2; ++mi) {  // A frag: lane=row, K chunks {hi*8, 16+hi*8}
            const u16* pp = as + (rw * 32 + mi * 16 + lrow) * 40 + hi * 8;
            Af[mi].u[0] = *(const u32x4*)(pp);
            Af[mi].u[1] = *(const u32x4*)(pp + 16);
        }
#pragma unroll
        for (int ni = 0; ni < 4; ++ni) {  // B frag: lane=col, K chunks {hi*8, 16+hi*8}
            const u16* pp = bs + (cw * 64 + ni * 16 + lrow) * 40 + hi * 8;
            Bf[ni].u[0] = *(const u32x4*)(pp);
            Bf[ni].u[1] = *(const u32x4*)(pp + 16);
        }
#pragma unroll
        for (int mi = 0; mi < 2; ++mi)
#pragma unroll
            for (int ni = 0; ni < 4; ++ni)
                C[mi][ni] = __builtin_amdgcn_wmma_f32_16x16x32_bf16(
                    false, asbf(Af[mi]), false, asbf(Bf[ni]), (short)0,
                    C[mi][ni], false, false);
    }

    // Epilogue. C layout: lane holds col = lrow; rows hi*8 + r in VGPR r.
#pragma unroll
    for (int mi = 0; mi < 2; ++mi) {
#pragma unroll
        for (int ni = 0; ni < 4; ++ni) {
            const int col = n0 + cw * 64 + ni * 16 + lrow;
#pragma unroll
            for (int r = 0; r < 8; ++r) {
                const int row = m0 + rw * 32 + mi * 16 + hi * 8 + r;
                float acc = C[mi][ni][r];
                if constexpr (EPI == EPI_QKV) {
                    const int which = col / Dm, c = col % Dm;
                    const int hh = c >> 6, dh = c & 63;
                    const int bb = row >> 10, ll = row & 1023;
                    const size_t bh = (size_t)bb * Hh + hh;
                    const u16 hv = f2bf(acc);
                    if (which == 0)      qOut[(bh * Lseq + ll) * DHd + dh] = hv;
                    else if (which == 1) kOut[(bh * Lseq + ll) * DHd + dh] = hv;
                    else                 vtOut[(bh * DHd + dh) * Lseq + ll] = hv; // v transposed
                } else if constexpr (EPI == EPI_PROJ) {
                    const size_t idx = (size_t)row * N + col;
                    fout[idx] = acc + bias[col] + resid[idx];        // x1 = x + proj
                } else if constexpr (EPI == EPI_FC1) {
                    const float t = acc + bias[col];
                    const float ge = 0.5f * t * (1.f + erff(t * 0.70710678118f));
                    hout[(size_t)row * N + col] = f2bf(ge);
                } else {                                             // EPI_FC2
                    const size_t idx = (size_t)row * N + col;
                    fout[idx] = acc + bias[col] + resid[idx];        // out = x1 + mlp
                }
            }
        }
    }
}

// ---------------------------------------------------------------------------
// Flash attention: grid (L/128, B*H). 8 waves; each wave owns 16 query rows,
// streams 64-key blocks with online softmax. K/V tiles are staged ONCE per
// block into LDS via double-buffered async DMA (waves share them); the P
// (C-layout -> A-layout) conversion uses wave-private LDS, which needs no
// barriers (per-wave LDS ops are in-order).
// Q,K bf16 [B,H,L,64]; V bf16 transposed [B,H,64,L]; out bf16 [B*L,768].
// ---------------------------------------------------------------------------
static constexpr int KT = 64 * 72;    // one 64x64 tile (+8 u16 row pad)

__global__ __launch_bounds__(256) void attn_kernel(const u16* __restrict__ Q,
                                                   const u16* __restrict__ Kt,
                                                   const u16* __restrict__ Vt,
                                                   u16* __restrict__ O) {
    __shared__ u16 Ks[2 * KT];
    __shared__ u16 Vs[2 * KT];
    __shared__ u16 Plds[8 * 16 * 72];
    const int tid = threadIdx.x;
    const int wave = tid >> 5, lane = tid & 31;
    const int lrow = lane & 15, hi = lane >> 4;
    const int bh = blockIdx.y, b = bh / Hh, h = bh % Hh;
    const u16* qp = Q  + (size_t)bh * Lseq * DHd;
    const u16* kp = Kt + (size_t)bh * Lseq * DHd;
    const u16* vp = Vt + (size_t)bh * DHd * Lseq;
    const int qRow0 = blockIdx.x * 128 + wave * 16;
    u16* pw = Plds + wave * 16 * 72;

    // stage loader: 64 rows x 128B; 4 threads/row x 32B each
    const int srow = tid >> 2, sseg = tid & 3;
    const u32 dK = lds_addr(Ks + srow * 72 + sseg * 16);
    const u32 dV = lds_addr(Vs + srow * 72 + sseg * 16);
    auto issue = [&](int j64, int buf) {
        async_copy32(dK + buf * (KT * 2), kp + (size_t)(j64 + srow) * DHd + sseg * 16);
        async_copy32(dV + buf * (KT * 2), vp + (size_t)srow * Lseq + j64 + sseg * 16);
    };

    Frag aQ[2];
    {
        const int ar = qRow0 + lrow, kc = hi * 8;
#pragma unroll
        for (int f = 0; f < 2; ++f) {
            const u16* p = qp + (size_t)ar * DHd + f * 32 + kc;
            aQ[f].u[0] = *(const u32x4*)(p);
            aQ[f].u[1] = *(const u32x4*)(p + 16);
        }
    }

    float m[8], l[8];
    v8f Oacc[4];
#pragma unroll
    for (int r = 0; r < 8; ++r) { m[r] = -1e30f; l[r] = 0.f; }
#pragma unroll
    for (int n = 0; n < 4; ++n) Oacc[n] = vzero();

    const float scale = 0.125f;  // dh^-0.5 = 1/8
    const int nj = Lseq / 64;
    issue(0, 0);
    for (int j = 0; j < nj; ++j) {
        const int p = j & 1;
        wait_async0();
        __syncthreads();
        if (j + 1 < nj) issue((j + 1) * 64, 1 - p);
        const u16* ks = Ks + p * KT;
        const u16* vs = Vs + p * KT;

        v8f S[4];
#pragma unroll
        for (int n = 0; n < 4; ++n) S[n] = vzero();
        // S = Q * K^T  (B operand rows = key rows of K, contiguous dh)
#pragma unroll
        for (int n = 0; n < 4; ++n) {
            const u16* pr = ks + (n * 16 + lrow) * 72 + hi * 8;
#pragma unroll
            for (int kk = 0; kk < 2; ++kk) {
                Frag bK;
                bK.u[0] = *(const u32x4*)(pr + kk * 32);
                bK.u[1] = *(const u32x4*)(pr + kk * 32 + 16);
                S[n] = __builtin_amdgcn_wmma_f32_16x16x32_bf16(
                    false, asbf(aQ[kk]), false, asbf(bK), (short)0, S[n], false, false);
            }
        }
#pragma unroll
        for (int n = 0; n < 4; ++n)
#pragma unroll
            for (int r = 0; r < 8; ++r) S[n][r] *= scale;

        // online softmax: row stats replicated across each 16-lane half
        float t[8];
#pragma unroll
        for (int r = 0; r < 8; ++r)
            t[r] = fmaxf(fmaxf(S[0][r], S[1][r]), fmaxf(S[2][r], S[3][r]));
#pragma unroll
        for (int mk = 1; mk < 16; mk <<= 1)
#pragma unroll
            for (int r = 0; r < 8; ++r) t[r] = fmaxf(t[r], __shfl_xor(t[r], mk, 32));
        float alpha[8];
#pragma unroll
        for (int r = 0; r < 8; ++r) {
            float mn = fmaxf(m[r], t[r]);
            alpha[r] = __expf(m[r] - mn);
            m[r] = mn;
        }
#pragma unroll
        for (int n = 0; n < 4; ++n)
#pragma unroll
            for (int r = 0; r < 8; ++r) S[n][r] = __expf(S[n][r] - m[r]);
#pragma unroll
        for (int r = 0; r < 8; ++r) t[r] = S[0][r] + S[1][r] + S[2][r] + S[3][r];
#pragma unroll
        for (int mk = 1; mk < 16; mk <<= 1)
#pragma unroll
            for (int r = 0; r < 8; ++r) t[r] += __shfl_xor(t[r], mk, 32);
#pragma unroll
        for (int r = 0; r < 8; ++r) l[r] = l[r] * alpha[r] + t[r];
#pragma unroll
        for (int n = 0; n < 4; ++n)
#pragma unroll
            for (int r = 0; r < 8; ++r) Oacc[n][r] *= alpha[r];

        // P: C-layout -> wave-private LDS -> A-layout (in-order per wave)
#pragma unroll
        for (int n = 0; n < 4; ++n) {
            const int col = n * 16 + lrow;
#pragma unroll
            for (int r = 0; r < 8; ++r) pw[(hi * 8 + r) * 72 + col] = f2bf(S[n][r]);
        }
        Frag aP[2];
#pragma unroll
        for (int f = 0; f < 2; ++f) {
            const u16* pp = pw + lrow * 72 + f * 32 + hi * 8;
            aP[f].u[0] = *(const u32x4*)(pp);
            aP[f].u[1] = *(const u32x4*)(pp + 16);
        }
        // O += P * V  (B operand rows = output dh cols of V^T, contiguous keys)
#pragma unroll
        for (int n = 0; n < 4; ++n) {
            const u16* pr = vs + (n * 16 + lrow) * 72 + hi * 8;
#pragma unroll
            for (int kk = 0; kk < 2; ++kk) {
                Frag bV;
                bV.u[0] = *(const u32x4*)(pr + kk * 32);
                bV.u[1] = *(const u32x4*)(pr + kk * 32 + 16);
                Oacc[n] = __builtin_amdgcn_wmma_f32_16x16x32_bf16(
                    false, asbf(aP[kk]), false, asbf(bV), (short)0, Oacc[n], false, false);
            }
        }
    }

    // normalize and store as [B*L, D] bf16 (A operand of proj GEMM)
#pragma unroll
    for (int n = 0; n < 4; ++n) {
        const int col = h * DHd + n * 16 + lrow;
#pragma unroll
        for (int r = 0; r < 8; ++r) {
            const int rowl = qRow0 + hi * 8 + r;
            O[((size_t)(b * Lseq + rowl)) * Dm + col] = f2bf(Oacc[n][r] / l[r]);
        }
    }
}

// ---------------------------------------------------------------------------
extern "C" void kernel_launch(void* const* d_in, const int* in_sizes, int n_in,
                              void* d_out, int out_size, void* d_ws, size_t ws_size,
                              hipStream_t stream) {
    const float* x      = (const float*)d_in[0];
    const float* qkv_w  = (const float*)d_in[1];
    const float* proj_w = (const float*)d_in[2];
    const float* proj_b = (const float*)d_in[3];
    const float* ln1_g  = (const float*)d_in[4];
    const float* ln1_b  = (const float*)d_in[5];
    const float* ln2_g  = (const float*)d_in[6];
    const float* ln2_b  = (const float*)d_in[7];
    const float* fc1_w  = (const float*)d_in[8];
    const float* fc1_b  = (const float*)d_in[9];
    const float* fc2_w  = (const float*)d_in[10];
    const float* fc2_b  = (const float*)d_in[11];
    float* out = (float*)d_out;

    char* ws = (char*)d_ws;
    size_t off = 0;
    auto alloc = [&](size_t bytes) -> char* {
        char* p = ws + off;
        off += (bytes + 255) & ~size_t(255);
        return p;
    };
    const size_t ML = (size_t)Bsz * Lseq;                 // 8192 rows
    u16*   wq_t  = (u16*)alloc((size_t)3 * Dm * Dm * 2);  // [2304][768]
    u16*   wp_t  = (u16*)alloc((size_t)Dm * Dm * 2);      // [768][768]
    u16*   wf1_t = (u16*)alloc((size_t)HIDd * Dm * 2);    // [3072][768]
    u16*   wf2_t = (u16*)alloc((size_t)Dm * HIDd * 2);    // [768][3072]
    u16*   h1    = (u16*)alloc(ML * Dm * 2);
    u16*   qb    = (u16*)alloc((size_t)Bsz * Hh * Lseq * DHd * 2);
    u16*   kb    = (u16*)alloc((size_t)Bsz * Hh * Lseq * DHd * 2);
    u16*   vtb   = (u16*)alloc((size_t)Bsz * Hh * DHd * Lseq * 2);
    u16*   ao    = (u16*)alloc(ML * Dm * 2);
    float* x1    = (float*)alloc(ML * Dm * 4);
    u16*   h2    = (u16*)alloc(ML * Dm * 2);
    u16*   h3    = (u16*)alloc(ML * HIDd * 2);

    // weights -> bf16, transposed [N][K]
    cvt_transpose_kernel<<<(Dm * 3 * Dm + 255) / 256, 256, 0, stream>>>(qkv_w, wq_t, Dm, 3 * Dm);
    cvt_transpose_kernel<<<(Dm * Dm + 255) / 256, 256, 0, stream>>>(proj_w, wp_t, Dm, Dm);
    cvt_transpose_kernel<<<(Dm * HIDd + 255) / 256, 256, 0, stream>>>(fc1_w, wf1_t, Dm, HIDd);
    cvt_transpose_kernel<<<(HIDd * Dm + 255) / 256, 256, 0, stream>>>(fc2_w, wf2_t, HIDd, Dm);

    // h1 = LN1(x)
    ln_kernel<<<(int)ML, 256, 0, stream>>>(x, ln1_g, ln1_b, h1);
    // qkv = h1 @ Wqkv, scattered into q, k, v^T
    gemm_bf16_kernel<EPI_QKV><<<dim3(3 * Dm / 128, ML / 128), 256, 0, stream>>>(
        h1, wq_t, (int)ML, 3 * Dm, Dm, nullptr, nullptr, nullptr, nullptr, qb, kb, vtb);
    // attention
    attn_kernel<<<dim3(Lseq / 128, Bsz * Hh), 256, 0, stream>>>(qb, kb, vtb, ao);
    // x1 = x + ao @ Wproj + b
    gemm_bf16_kernel<EPI_PROJ><<<dim3(Dm / 128, ML / 128), 256, 0, stream>>>(
        ao, wp_t, (int)ML, Dm, Dm, proj_b, x, x1, nullptr, nullptr, nullptr, nullptr);
    // h2 = LN2(x1)
    ln_kernel<<<(int)ML, 256, 0, stream>>>(x1, ln2_g, ln2_b, h2);
    // h3 = gelu(h2 @ Wfc1 + b)
    gemm_bf16_kernel<EPI_FC1><<<dim3(HIDd / 128, ML / 128), 256, 0, stream>>>(
        h2, wf1_t, (int)ML, HIDd, Dm, fc1_b, nullptr, nullptr, h3, nullptr, nullptr, nullptr);
    // out = x1 + h3 @ Wfc2 + b
    gemm_bf16_kernel<EPI_FC2><<<dim3(Dm / 128, ML / 128), 256, 0, stream>>>(
        h3, wf2_t, (int)ML, Dm, HIDd, fc2_b, x1, out, nullptr, nullptr, nullptr, nullptr);

    (void)in_sizes; (void)n_in; (void)out_size; (void)ws_size;
}
